// Net_58969900974293
// MI455X (gfx1250) — matrix operand order repaired and checked by reading, not compiled, not run on hardware
//
#include <hip/hip_runtime.h>
#include <hip/hip_bf16.h>

typedef __attribute__((ext_vector_type(16))) _Float16 v16h;
typedef __attribute__((ext_vector_type(8)))  _Float16 v8h;
typedef __attribute__((ext_vector_type(8)))  float    v8f;

#define IN_CH  128
#define HEADS  4
#define OUT_CH 32
#define HC     128   // HEADS*OUT_CH
#define NEG_SLOPE 0.2f
#define MT 64        // GEMM M-tile rows per block

// ---------------------------------------------------------------------------
// Kernel 1: split W (f32 [K=128][N=128]) into transposed f16 hi/lo: Wt[n][k]
// ---------------------------------------------------------------------------
__global__ void wt_split_kernel(const float* __restrict__ W,
                                _Float16* __restrict__ wth,
                                _Float16* __restrict__ wtl) {
    int i = blockIdx.x * blockDim.x + threadIdx.x;
    if (i >= 128 * 128) return;
    int n = i >> 7;
    int k = i & 127;
    float w = W[k * 128 + n];
    _Float16 hi = (_Float16)w;
    _Float16 lo = (_Float16)(w - (float)hi);
    wth[n * 128 + k] = hi;
    wtl[n * 128 + k] = lo;
}

// ---------------------------------------------------------------------------
// Kernel 2: xw = x @ W via split-f16 WMMA (hi*hi + hi*lo + lo*hi), f32 accum.
// Block: 256 threads (8 waves) -> one 64x128 output tile.
// Wave w owns N-tile w; iterates 4 M-tiles of 16.
// Full blocks (all but the last) take a branchless path.
// ---------------------------------------------------------------------------
__global__ __launch_bounds__(256) void gemm_kernel(
    const float* __restrict__ x,
    const _Float16* __restrict__ wth,
    const _Float16* __restrict__ wtl,
    float* __restrict__ xw, int N) {
    __shared__ _Float16 Ah[MT][136];   // +8 halves pad: conflict-free frag reads
    __shared__ _Float16 Al[MT][136];

    const int  tid  = threadIdx.x;
    const int  row0 = blockIdx.x * MT;
    const bool full = (row0 + MT) <= N;   // block-uniform

    // Stage A tile: MT x 128 f32 -> split f16 hi/lo in LDS (float4 loads)
    for (int i = tid; i < MT * 32; i += 256) {
        int r  = i >> 5;
        int c4 = (i & 31) << 2;
        float4 v;
        if (full) {
            v = *(const float4*)(x + (size_t)(row0 + r) * IN_CH + c4);
        } else {
            v = make_float4(0.f, 0.f, 0.f, 0.f);
            if (row0 + r < N)
                v = *(const float4*)(x + (size_t)(row0 + r) * IN_CH + c4);
        }
        float vv[4] = {v.x, v.y, v.z, v.w};
#pragma unroll
        for (int j = 0; j < 4; ++j) {
            _Float16 h = (_Float16)vv[j];
            Ah[r][c4 + j] = h;
            Al[r][c4 + j] = (_Float16)(vv[j] - (float)h);
        }
    }
    __syncthreads();

    const int wv    = tid >> 5;        // wave id = N-tile index (0..7)
    const int lane  = tid & 31;
    const int lrow  = lane & 15;
    const int lhalf = lane >> 4;
    const int ncol  = wv * 16 + lrow;  // global output column for B frags / C

    for (int mt = 0; mt < MT / 16; ++mt) {
        v8f acc = {};
#pragma unroll
        for (int kt = 0; kt < 4; ++kt) {
            const int kbase = kt * 32;
            // A fragment (16x32 f16): lane holds 16 contiguous k at
            // k=kbase+16*lhalf (matches B).
            const _Float16* pah = &Ah[mt * 16 + lrow][kbase + 16 * lhalf];
            const _Float16* pal = &Al[mt * 16 + lrow][kbase + 16 * lhalf];
            v8h ah0 = *(const v8h*)(pah);
            v8h ah1 = *(const v8h*)(pah + 8);
            v8h al0 = *(const v8h*)(pal);
            v8h al1 = *(const v8h*)(pal + 8);
            v16h ah, al;
#pragma unroll
            for (int j = 0; j < 8; ++j) {
                ah[j] = ah0[j]; ah[8 + j] = ah1[j];
                al[j] = al0[j]; al[8 + j] = al1[j];
            }
            // B fragment (32x16): lane holds 16 contiguous k for column ncol.
            const _Float16* pbh = wth + (size_t)ncol * 128 + kbase + 16 * lhalf;
            const _Float16* pbl = wtl + (size_t)ncol * 128 + kbase + 16 * lhalf;
            v16h bh = *(const v16h*)pbh;
            v16h bl = *(const v16h*)pbl;

            acc = __builtin_amdgcn_wmma_f32_16x16x32_f16(
                false, ah, false, bh, (short)0, acc, false, false);
            acc = __builtin_amdgcn_wmma_f32_16x16x32_f16(
                false, ah, false, bl, (short)0, acc, false, false);
            acc = __builtin_amdgcn_wmma_f32_16x16x32_f16(
                false, al, false, bh, (short)0, acc, false, false);
        }
        // C/D layout: VGPR v -> M = v + 8*lhalf, N = lrow
        const int mbase = row0 + mt * 16 + 8 * lhalf;
        float* po = xw + (size_t)mbase * HC + ncol;
        if (full) {
#pragma unroll
            for (int v = 0; v < 8; ++v) po[(size_t)v * HC] = acc[v];
        } else {
#pragma unroll
            for (int v = 0; v < 8; ++v)
                if (mbase + v < N) po[(size_t)v * HC] = acc[v];
        }
    }
}

// ---------------------------------------------------------------------------
// Kernel 3: per-node attention coefficients. One wave per node.
// lane owns 4 channels; head h = lane>>3; 8-lane shuffle reduction.
// ---------------------------------------------------------------------------
__global__ __launch_bounds__(256) void attn_kernel(
    const float* __restrict__ xw,
    const float* __restrict__ att_src,
    const float* __restrict__ att_dst,
    float* __restrict__ a_src, float* __restrict__ a_dst, int N) {
    const int wv   = threadIdx.x >> 5;
    const int lane = threadIdx.x & 31;
    const int node = blockIdx.x * 8 + wv;
    if (node >= N) return;
    const int h  = lane >> 3;
    const int co = lane << 2;
    float4 v = *(const float4*)(xw + (size_t)node * HC + co);
    const float* as = att_src + h * OUT_CH + (lane & 7) * 4;
    const float* ad = att_dst + h * OUT_CH + (lane & 7) * 4;
    float ss = v.x * as[0] + v.y * as[1] + v.z * as[2] + v.w * as[3];
    float sd = v.x * ad[0] + v.y * ad[1] + v.z * ad[2] + v.w * ad[3];
#pragma unroll
    for (int off = 4; off; off >>= 1) {
        ss += __shfl_xor(ss, off, 32);
        sd += __shfl_xor(sd, off, 32);
    }
    if ((lane & 7) == 0) {
        a_src[node * HEADS + h] = ss;
        a_dst[node * HEADS + h] = sd;
    }
}

// ---------------------------------------------------------------------------
// Kernel 4: init out=bias, segmax=-inf-ish, denom=0
// ---------------------------------------------------------------------------
__global__ void init_kernel(float* __restrict__ out, const float* __restrict__ bias,
                            float* __restrict__ segmax, float* __restrict__ denom,
                            int N) {
    int i = blockIdx.x * blockDim.x + threadIdx.x;
    if (i < N * HC) out[i] = bias[i & (HC - 1)];
    if (i < N * HEADS) { segmax[i] = -3.0e38f; denom[i] = 0.0f; }
}

// ---------------------------------------------------------------------------
// Kernel 5: edge logits (leaky relu) + segmented atomic max over dst.
// idx >= E means self-loop node (idx-E).
// ---------------------------------------------------------------------------
__global__ void logit_max_kernel(const int* __restrict__ src, const int* __restrict__ dst,
                                 const float* __restrict__ a_src, const float* __restrict__ a_dst,
                                 float* __restrict__ ebuf, float* __restrict__ segmax,
                                 int E, int Etot) {
    int idx = blockIdx.x * blockDim.x + threadIdx.x;
    if (idx >= Etot) return;
    int s, d;
    if (idx < E) { s = src[idx]; d = dst[idx]; } else { s = d = idx - E; }
    float4 as = *(const float4*)(a_src + (size_t)s * HEADS);
    float4 ad = *(const float4*)(a_dst + (size_t)d * HEADS);
    float l[4] = {as.x + ad.x, as.y + ad.y, as.z + ad.z, as.w + ad.w};
#pragma unroll
    for (int h = 0; h < HEADS; ++h) {
        l[h] = (l[h] > 0.f) ? l[h] : NEG_SLOPE * l[h];
        __hip_atomic_fetch_max(&segmax[(size_t)d * HEADS + h], l[h],
                               __ATOMIC_RELAXED, __HIP_MEMORY_SCOPE_AGENT);
    }
    *(float4*)(ebuf + (size_t)idx * HEADS) = make_float4(l[0], l[1], l[2], l[3]);
}

// ---------------------------------------------------------------------------
// Kernel 6: e = exp(l - max[dst]); denom[dst] += e
// ---------------------------------------------------------------------------
__global__ void expsum_kernel(const int* __restrict__ dst,
                              float* __restrict__ ebuf,
                              const float* __restrict__ segmax,
                              float* __restrict__ denom,
                              int E, int Etot) {
    int idx = blockIdx.x * blockDim.x + threadIdx.x;
    if (idx >= Etot) return;
    int d = (idx < E) ? dst[idx] : idx - E;
    float4 l = *(const float4*)(ebuf + (size_t)idx * HEADS);
    float4 m = *(const float4*)(segmax + (size_t)d * HEADS);
    float e0 = __expf(l.x - m.x), e1 = __expf(l.y - m.y);
    float e2 = __expf(l.z - m.z), e3 = __expf(l.w - m.w);
    *(float4*)(ebuf + (size_t)idx * HEADS) = make_float4(e0, e1, e2, e3);
    atomicAdd(&denom[(size_t)d * HEADS + 0], e0);
    atomicAdd(&denom[(size_t)d * HEADS + 1], e1);
    atomicAdd(&denom[(size_t)d * HEADS + 2], e2);
    atomicAdd(&denom[(size_t)d * HEADS + 3], e3);
}

// ---------------------------------------------------------------------------
// Kernel 7: out[dst] += xw[src] * alpha. One wave per edge; lane owns 4 chans.
// xw and out are L2-resident (51 MB each, 192 MB L2).
// ---------------------------------------------------------------------------
__global__ __launch_bounds__(256) void agg_kernel(
    const int* __restrict__ src, const int* __restrict__ dst,
    const float* __restrict__ xw, const float* __restrict__ ebuf,
    const float* __restrict__ denom, float* __restrict__ out,
    int E, int Etot) {
    const int wv   = threadIdx.x >> 5;
    const int lane = threadIdx.x & 31;
    const int idx  = blockIdx.x * 8 + wv;
    if (idx >= Etot) return;
    int s, d;
    if (idx < E) { s = src[idx]; d = dst[idx]; } else { s = d = idx - E; }
    const int h = lane >> 3;
    float alpha = ebuf[(size_t)idx * HEADS + h] / denom[(size_t)d * HEADS + h];
    float4 v = *(const float4*)(xw + (size_t)s * HC + lane * 4);
    float* o = out + (size_t)d * HC + lane * 4;
    atomicAdd(o + 0, v.x * alpha);
    atomicAdd(o + 1, v.y * alpha);
    atomicAdd(o + 2, v.z * alpha);
    atomicAdd(o + 3, v.w * alpha);
}

// ---------------------------------------------------------------------------
extern "C" void kernel_launch(void* const* d_in, const int* in_sizes, int n_in,
                              void* d_out, int out_size, void* d_ws, size_t ws_size,
                              hipStream_t stream) {
    const float* x        = (const float*)d_in[0];
    const int*   ei       = (const int*)d_in[1];
    const float* W        = (const float*)d_in[2];
    const float* att_src  = (const float*)d_in[3];
    const float* att_dst  = (const float*)d_in[4];
    const float* bias     = (const float*)d_in[5];

    const int N = in_sizes[0] / IN_CH;
    const int E = in_sizes[1] / 2;
    const int Etot = E + N;
    const int* src = ei;
    const int* dst = ei + E;

    // Workspace layout (floats)
    float* xw     = (float*)d_ws;
    float* a_src  = xw + (size_t)N * HC;
    float* a_dst  = a_src + (size_t)N * HEADS;
    float* segmax = a_dst + (size_t)N * HEADS;
    float* denom  = segmax + (size_t)N * HEADS;
    float* ebuf   = denom + (size_t)N * HEADS;
    _Float16* wth = (_Float16*)(ebuf + (size_t)Etot * HEADS);
    _Float16* wtl = wth + 128 * 128;

    float* out = (float*)d_out;

    wt_split_kernel<<<(128 * 128 + 255) / 256, 256, 0, stream>>>(W, wth, wtl);
    gemm_kernel<<<(N + MT - 1) / MT, 256, 0, stream>>>(x, wth, wtl, xw, N);
    attn_kernel<<<(N + 7) / 8, 256, 0, stream>>>(xw, att_src, att_dst, a_src, a_dst, N);
    init_kernel<<<((size_t)N * HC + 255) / 256, 256, 0, stream>>>(out, bias, segmax, denom, N);
    logit_max_kernel<<<(Etot + 255) / 256, 256, 0, stream>>>(src, dst, a_src, a_dst,
                                                             ebuf, segmax, E, Etot);
    expsum_kernel<<<(Etot + 255) / 256, 256, 0, stream>>>(dst, ebuf, segmax, denom, E, Etot);
    agg_kernel<<<(Etot + 7) / 8, 256, 0, stream>>>(src, dst, xw, ebuf, denom, out, E, Etot);
}
